// SelfAttention_12713103197312
// MI455X (gfx1250) — matrix operand order repaired and checked by reading, not compiled, not run on hardware
//
#include <hip/hip_runtime.h>
#include <hip/hip_bf16.h>
#include <stdint.h>

// MI455X / gfx1250, wave32. WMMA bf16 + Tensor Data Mover staging.

#if __has_include(<hip/amd_detail/amd_gfx1250_TDM.h>)
#define GFX1250_TDM_6ARG 1
#endif

typedef __attribute__((ext_vector_type(16))) __bf16 v16bf;
typedef __attribute__((ext_vector_type(8)))  float  v8f;
typedef __attribute__((ext_vector_type(4)))  unsigned int v4u;
typedef __attribute__((ext_vector_type(8)))  int v8i;
typedef __attribute__((ext_vector_type(4)))  int v4i;

#define EMBED 1024
#define HEADS 16
#define HDIM  64
#define SEQ   2048
#define NBATCH 4

union FragBF { v16bf v; uint4 u[2]; };
union Pack8  { unsigned short s8[8]; uint4 v; };

__device__ __forceinline__ unsigned short f32_bf16(float f) {
  union { float f; uint32_t u; } c; c.f = f;
  uint32_t u = c.u;
  u += 0x7FFFu + ((u >> 16) & 1u);   // RNE
  return (unsigned short)(u >> 16);
}

__device__ __forceinline__ v8f wmma_bf16(const FragBF& a, const FragBF& b, v8f c) {
  return __builtin_amdgcn_wmma_f32_16x16x32_bf16(false, a.v, false, b.v,
                                                 (short)0, c, false, false);
}

// ---------------------------------------------------------------------------
// TDM: DMA a 32(k) x 128(rows) bf16 tile, row-major stride `stride0` elems,
// into LDS with 16 B padding every 64 B row -> 80 B LDS row stride
// (20-dword bank stride, conflict-free for 16-lane fragment reads).
// D# fields per ISA ch.8: group0 = {count, lds_addr, global_addr, type=2},
// group1 = {data_size=2B, pad_enable, pad_interval=16dw, pad_amount=4dw,
//           tensor_dim0/1, tile_dim0=32, tile_dim1=128, dim0_stride}.
// ---------------------------------------------------------------------------
__device__ __forceinline__ void tdm_load_tile_2d(
    unsigned lds_byte_addr, unsigned long long gaddr,
    unsigned tensor_d0, unsigned tensor_d1, unsigned stride0)
{
  v4u g0;
  g0.x = 1u;                                              // count=1, user D#
  g0.y = lds_byte_addr;
  g0.z = (unsigned)gaddr;
  g0.w = (unsigned)((gaddr >> 32) & 0x01FFFFFFull) | (2u << 30);  // type=2
  v8i g1;
  g1[0] = (int)((1u << 16) | (1u << 20) | (3u << 22) | (3u << 25));
  g1[1] = (int)((tensor_d0 & 0xFFFFu) << 16);
  g1[2] = (int)((tensor_d0 >> 16) | ((tensor_d1 & 0xFFFFu) << 16));
  g1[3] = (int)((tensor_d1 >> 16) | (32u << 16));         // tile_dim0 = 32
  g1[4] = (int)128u;                                      // tile_dim1 = 128
  g1[5] = (int)stride0;
  g1[6] = 0;
  g1[7] = 0;
  v4i z4 = {0, 0, 0, 0};
#ifdef GFX1250_TDM_6ARG
  v8i z8 = {0, 0, 0, 0, 0, 0, 0, 0};
  __builtin_amdgcn_tensor_load_to_lds(g0, g1, z4, z4, z8, 0);
#else
  __builtin_amdgcn_tensor_load_to_lds(g0, g1, z4, z4, 0);
#endif
}

// ---------------------------------------------------------------------------
// f32 -> bf16 bulk convert (x and the four weight matrices), 8 elems/thread.
// ---------------------------------------------------------------------------
__global__ __launch_bounds__(256) void cvt_f32_bf16_kernel(
    const float* __restrict__ src, unsigned short* __restrict__ dst, int n8)
{
  int i = blockIdx.x * 256 + threadIdx.x;
  if (i >= n8) return;
  const float4* s = (const float4*)src + (size_t)i * 2;
  float4 a = s[0], b = s[1];
  Pack8 p;
  p.s8[0] = f32_bf16(a.x); p.s8[1] = f32_bf16(a.y);
  p.s8[2] = f32_bf16(a.z); p.s8[3] = f32_bf16(a.w);
  p.s8[4] = f32_bf16(b.x); p.s8[5] = f32_bf16(b.y);
  p.s8[6] = f32_bf16(b.z); p.s8[7] = f32_bf16(b.w);
  *(uint4*)(dst + (size_t)i * 8) = p.v;
}

// ---------------------------------------------------------------------------
// GEMM: C[Mrows x 1024] = A[Mrows x 1024] * W^T, bf16 in / f32 accum.
// TDM double-buffered staging; 8 waves, each 2x4 WMMA tiles; block 128x128.
// Orientation WA=true : A-operand = W  -> lane accum holds 8 consecutive
//                       output dims   -> packed b128 stores ([N,H,S,D] / f32)
// Orientation WA=false: A-operand = x  -> lane accum holds 8 consecutive s
//                       -> packed b128 stores for V^T [N,H,D,S]
// mode 0: Q bf16 [N,H,S,D] * 0.125 | mode 1: K | mode 2: V^T | mode 3: f32+bias
// ---------------------------------------------------------------------------
template<bool WA>
__global__ __launch_bounds__(256) void gemm_tdm_kernel(
    const unsigned short* __restrict__ Arows, const unsigned short* __restrict__ Wb,
    int Mrows, unsigned short* __restrict__ outB, float* __restrict__ outF,
    const float* __restrict__ bias, int mode)
{
  __shared__ alignas(16) unsigned short As[2][128 * 40];
  __shared__ alignas(16) unsigned short Bs[2][128 * 40];

  const int t    = threadIdx.x;
  const int lane = t & 31;
  const int w    = t >> 5;
  const int wq   = w >> 1;        // 0..3 : 32-wide strip on A-operand dim
  const int wh   = w & 1;         // 0..1 : 64-wide strip on B-operand dim
  const int g    = lane >> 4;
  const int l16  = lane & 15;
  const int bm   = blockIdx.y * 128;   // rows of A
  const int bn   = blockIdx.x * 128;   // weight output dims

  const unsigned long long abase = (unsigned long long)(uintptr_t)Arows;
  const unsigned long long wbase = (unsigned long long)(uintptr_t)Wb;
  const unsigned ldsA[2] = { (unsigned)(uintptr_t)&As[0][0], (unsigned)(uintptr_t)&As[1][0] };
  const unsigned ldsB[2] = { (unsigned)(uintptr_t)&Bs[0][0], (unsigned)(uintptr_t)&Bs[1][0] };

  v8f zero = {0.f,0.f,0.f,0.f,0.f,0.f,0.f,0.f};
  v8f acc[2][4];
  #pragma unroll
  for (int i = 0; i < 2; i++)
    #pragma unroll
    for (int j = 0; j < 4; j++) acc[i][j] = zero;

  if (t < 32) {     // wave 0 drives the Tensor Data Mover
    tdm_load_tile_2d(ldsA[0], abase + (unsigned long long)bm * EMBED * 2,
                     EMBED, (unsigned)Mrows, EMBED);
    tdm_load_tile_2d(ldsB[0], wbase + (unsigned long long)bn * EMBED * 2,
                     EMBED, EMBED, EMBED);
    __builtin_amdgcn_s_wait_tensorcnt(0);
  }
  __syncthreads();

  int cur = 0;
  for (int k0 = 0; k0 < EMBED; k0 += 32) {
    if (k0 + 32 < EMBED && t < 32) {   // async-load next slab into other buffer
      int nxt = cur ^ 1;
      tdm_load_tile_2d(ldsA[nxt],
                       abase + ((unsigned long long)bm * EMBED + k0 + 32) * 2,
                       EMBED, (unsigned)Mrows, EMBED);
      tdm_load_tile_2d(ldsB[nxt],
                       wbase + ((unsigned long long)bn * EMBED + k0 + 32) * 2,
                       EMBED, EMBED, EMBED);
    }
    const unsigned short* sa = As[cur];
    const unsigned short* sb = Bs[cur];

    FragBF afr[2], bfr[4];
    #pragma unroll
    for (int mt = 0; mt < 2; mt++) {
      const unsigned short* src = WA ? sb : sa;
      int arow = wq * 32 + mt * 16 + l16;
      afr[mt].u[0] = *(const uint4*)&src[arow * 40 + g * 8];       // K g*8..+7
      afr[mt].u[1] = *(const uint4*)&src[arow * 40 + 16 + g * 8];  // K 16+g*8..
    }
    #pragma unroll
    for (int nt = 0; nt < 4; nt++) {
      const unsigned short* src = WA ? sa : sb;
      int brow = wh * 64 + nt * 16 + l16;
      bfr[nt].u[0] = *(const uint4*)&src[brow * 40 + g * 8];       // K g*8..+7
      bfr[nt].u[1] = *(const uint4*)&src[brow * 40 + 16 + g * 8];
    }
    #pragma unroll
    for (int mt = 0; mt < 2; mt++)
      #pragma unroll
      for (int nt = 0; nt < 4; nt++)
        acc[mt][nt] = wmma_bf16(afr[mt], bfr[nt], acc[mt][nt]);

    if (t < 32) __builtin_amdgcn_s_wait_tensorcnt(0);
    __syncthreads();
    cur ^= 1;
  }

  // ---- epilogue: one b128 store per 16x16 tile per lane ----
  #pragma unroll
  for (int mt = 0; mt < 2; mt++)
    #pragma unroll
    for (int nt = 0; nt < 4; nt++) {
      if (WA) {
        int srow  = bm + wh * 64 + nt * 16 + l16;        // (n, s)
        int dbase = bn + wq * 32 + mt * 16 + g * 8;      // 8 consecutive dims
        if (mode == 3) {
          float4 b0 = *(const float4*)&bias[dbase];
          float4 b1 = *(const float4*)&bias[dbase + 4];
          float4 o0 = { acc[mt][nt][0] + b0.x, acc[mt][nt][1] + b0.y,
                        acc[mt][nt][2] + b0.z, acc[mt][nt][3] + b0.w };
          float4 o1 = { acc[mt][nt][4] + b1.x, acc[mt][nt][5] + b1.y,
                        acc[mt][nt][6] + b1.z, acc[mt][nt][7] + b1.w };
          *(float4*)&outF[(size_t)srow * EMBED + dbase]     = o0;
          *(float4*)&outF[(size_t)srow * EMBED + dbase + 4] = o1;
        } else {
          int n = srow >> 11, s = srow & 2047;
          int h = dbase >> 6, dd = dbase & 63;
          float sc = (mode == 0) ? 0.125f : 1.0f;
          Pack8 p;
          #pragma unroll
          for (int r = 0; r < 8; r++) p.s8[r] = f32_bf16(acc[mt][nt][r] * sc);
          *(uint4*)&outB[(((size_t)(n * HEADS + h)) * SEQ + s) * HDIM + dd] = p.v;
        }
      } else {   // V^T [N,H,D,S]: lane holds 8 consecutive s values
        int grow0 = bm + wq * 32 + mt * 16 + g * 8;      // first s of the 8
        int gcol  = bn + wh * 64 + nt * 16 + l16;        // fixed dim
        int n = grow0 >> 11, s0 = grow0 & 2047;
        int h = gcol >> 6, dd = gcol & 63;
        Pack8 p;
        #pragma unroll
        for (int r = 0; r < 8; r++) p.s8[r] = f32_bf16(acc[mt][nt][r]);
        *(uint4*)&outB[(((size_t)(n * HEADS + h)) * HDIM + dd) * SEQ + s0] = p.v;
      }
    }
}

// ---------------------------------------------------------------------------
// Flash attention: grid (S/64, H, N), 128 threads = 4 waves, 16 query rows
// per wave, 32 keys per step, online softmax. 1/sqrt(D) folded into Q.
// ---------------------------------------------------------------------------
__global__ __launch_bounds__(128) void flash_attn_kernel(
    const unsigned short* __restrict__ Q,
    const unsigned short* __restrict__ K,
    const unsigned short* __restrict__ Vt,
    unsigned short* __restrict__ O)
{
  __shared__ alignas(16) unsigned short plds[4][16 * 40];  // per-wave P scratch

  const int t    = threadIdx.x;
  const int lane = t & 31;
  const int w    = t >> 5;
  const int g    = lane >> 4;
  const int l16  = lane & 15;
  const int h    = blockIdx.y;
  const int n    = blockIdx.z;
  const int qbase = blockIdx.x * 64 + w * 16;

  const size_t headQK = ((size_t)(n * HEADS + h)) * SEQ * HDIM;
  const unsigned short* Qh = Q  + headQK;
  const unsigned short* Kh = K  + headQK;
  const unsigned short* Vh = Vt + ((size_t)(n * HEADS + h)) * HDIM * SEQ;

  FragBF qa[2];
  {
    const unsigned short* qrow = Qh + (size_t)(qbase + l16) * HDIM;
    qa[0].u[0] = *(const uint4*)(qrow + g * 8);
    qa[0].u[1] = *(const uint4*)(qrow + 16 + g * 8);
    qa[1].u[0] = *(const uint4*)(qrow + 32 + g * 8);
    qa[1].u[1] = *(const uint4*)(qrow + 48 + g * 8);
  }

  v8f zero = {0.f,0.f,0.f,0.f,0.f,0.f,0.f,0.f};
  v8f accO[4];
  float m_i[8], l_i[8];
  #pragma unroll
  for (int i = 0; i < 4; i++) accO[i] = zero;
  #pragma unroll
  for (int r = 0; r < 8; r++) { m_i[r] = -__builtin_inff(); l_i[r] = 0.f; }

  unsigned short* pw = plds[w];

  for (int jb = 0; jb < SEQ; jb += 32) {
    if (jb + 32 < SEQ) {
      __builtin_prefetch(Kh + (size_t)(jb + 32 + l16) * HDIM, 0, 1);
      __builtin_prefetch(Vh + (size_t)l16 * SEQ + jb + 32, 0, 1);
    }
    FragBF kb[2][2];
    #pragma unroll
    for (int kt = 0; kt < 2; kt++) {
      const unsigned short* krow = Kh + (size_t)(jb + kt * 16 + l16) * HDIM;
      #pragma unroll
      for (int dh = 0; dh < 2; dh++) {
        kb[kt][dh].u[0] = *(const uint4*)(krow + dh * 32 + g * 8);
        kb[kt][dh].u[1] = *(const uint4*)(krow + dh * 32 + 16 + g * 8);
      }
    }
    v8f st[2];
    #pragma unroll
    for (int kt = 0; kt < 2; kt++) {
      v8f s = zero;
      s = wmma_bf16(qa[0], kb[kt][0], s);
      s = wmma_bf16(qa[1], kb[kt][1], s);
      st[kt] = s;
    }
    #pragma unroll
    for (int r = 0; r < 8; r++) {
      float s0 = st[0][r], s1 = st[1][r];
      float mx = fmaxf(s0, s1);
      mx = fmaxf(mx, __shfl_xor(mx, 1));
      mx = fmaxf(mx, __shfl_xor(mx, 2));
      mx = fmaxf(mx, __shfl_xor(mx, 4));
      mx = fmaxf(mx, __shfl_xor(mx, 8));
      float mnew = fmaxf(m_i[r], mx);
      float p0 = __expf(s0 - mnew), p1 = __expf(s1 - mnew);
      float sum = p0 + p1;
      sum += __shfl_xor(sum, 1);
      sum += __shfl_xor(sum, 2);
      sum += __shfl_xor(sum, 4);
      sum += __shfl_xor(sum, 8);
      float alpha = __expf(m_i[r] - mnew);
      l_i[r] = l_i[r] * alpha + sum;
      m_i[r] = mnew;
      #pragma unroll
      for (int vt = 0; vt < 4; vt++) accO[vt][r] *= alpha;
      int prow = g * 8 + r;
      pw[prow * 40 + l16]      = f32_bf16(p0);
      pw[prow * 40 + 16 + l16] = f32_bf16(p1);
    }
    asm volatile("s_wait_dscnt 0x0" ::: "memory");   // cross-lane LDS RAW
    FragBF pa;
    pa.u[0] = *(const uint4*)&pw[l16 * 40 + g * 8];
    pa.u[1] = *(const uint4*)&pw[l16 * 40 + 16 + g * 8];
    #pragma unroll
    for (int vt = 0; vt < 4; vt++) {
      FragBF vb;
      const unsigned short* vrow = Vh + (size_t)(vt * 16 + l16) * SEQ + jb;
      vb.u[0] = *(const uint4*)(vrow + g * 8);
      vb.u[1] = *(const uint4*)(vrow + 16 + g * 8);
      accO[vt] = wmma_bf16(pa, vb, accO[vt]);
    }
  }

  #pragma unroll
  for (int vt = 0; vt < 4; vt++)
    #pragma unroll
    for (int r = 0; r < 8; r++) {
      int qrow = qbase + g * 8 + r;
      int col  = h * HDIM + vt * 16 + l16;
      O[(size_t)(n * SEQ + qrow) * EMBED + col] = f32_bf16(accO[vt][r] / l_i[r]);
    }
}

// ---------------------------------------------------------------------------
extern "C" void kernel_launch(void* const* d_in, const int* in_sizes, int n_in,
                              void* d_out, int out_size, void* d_ws, size_t ws_size,
                              hipStream_t stream) {
  (void)in_sizes; (void)n_in; (void)out_size; (void)ws_size;
  const float* x  = (const float*)d_in[0];
  const float* Wq = (const float*)d_in[1];
  const float* Wk = (const float*)d_in[2];
  const float* Wv = (const float*)d_in[3];
  const float* Wo = (const float*)d_in[4];
  const float* bo = (const float*)d_in[5];
  float* out = (float*)d_out;

  char* ws = (char*)d_ws;
  const size_t MB = (size_t)1024 * 1024;
  unsigned short* xb  = (unsigned short*)(ws);            // [8192,1024] bf16
  unsigned short* Wqb = (unsigned short*)(ws + 16 * MB);
  unsigned short* Wkb = (unsigned short*)(ws + 18 * MB);
  unsigned short* Wvb = (unsigned short*)(ws + 20 * MB);
  unsigned short* Wob = (unsigned short*)(ws + 22 * MB);
  unsigned short* Qb  = (unsigned short*)(ws + 24 * MB);  // [N,H,S,D]
  unsigned short* Kb  = (unsigned short*)(ws + 40 * MB);  // [N,H,S,D]
  unsigned short* Vtb = (unsigned short*)(ws + 56 * MB);  // [N,H,D,S]
  unsigned short* Ob  = (unsigned short*)(ws + 72 * MB);  // [N,S,H*D]

  const int MROWS = NBATCH * SEQ;  // 8192
  cvt_f32_bf16_kernel<<<(MROWS * EMBED / 8) / 256, 256, 0, stream>>>(x, xb, MROWS * EMBED / 8);
  cvt_f32_bf16_kernel<<<(EMBED * EMBED / 8) / 256, 256, 0, stream>>>(Wq, Wqb, EMBED * EMBED / 8);
  cvt_f32_bf16_kernel<<<(EMBED * EMBED / 8) / 256, 256, 0, stream>>>(Wk, Wkb, EMBED * EMBED / 8);
  cvt_f32_bf16_kernel<<<(EMBED * EMBED / 8) / 256, 256, 0, stream>>>(Wv, Wvb, EMBED * EMBED / 8);
  cvt_f32_bf16_kernel<<<(EMBED * EMBED / 8) / 256, 256, 0, stream>>>(Wo, Wob, EMBED * EMBED / 8);

  dim3 gg(EMBED / 128, MROWS / 128);   // (8, 64)
  gemm_tdm_kernel<true ><<<gg, 256, 0, stream>>>(xb, Wqb, MROWS, Qb,  nullptr, nullptr, 0);
  gemm_tdm_kernel<true ><<<gg, 256, 0, stream>>>(xb, Wkb, MROWS, Kb,  nullptr, nullptr, 1);
  gemm_tdm_kernel<false><<<gg, 256, 0, stream>>>(xb, Wvb, MROWS, Vtb, nullptr, nullptr, 2);

  flash_attn_kernel<<<dim3(SEQ / 64, HEADS, NBATCH), 128, 0, stream>>>(Qb, Kb, Vtb, Ob);

  gemm_tdm_kernel<true ><<<gg, 256, 0, stream>>>(Ob, Wob, MROWS, nullptr, out, bo, 3);
}
